// VectorQuantizer_89850715832791
// MI455X (gfx1250) — matrix operand order repaired and checked by reading, not compiled, not run on hardware
//
#include <hip/hip_runtime.h>

// CDNA5 / gfx1250 WMMA types
typedef __attribute__((ext_vector_type(16))) __bf16 v16bf;
typedef __attribute__((ext_vector_type(8)))  float  v8f;

#define EMB_DIM        64
#define NUM_CODES      1024
#define CHUNK          256   // codebook rows staged in LDS per pass
#define ROW_PAD        72    // bf16 elems per LDS row (64 + 8 pad -> 144B, conflict-free)
#define WAVES_PER_BLK  8
#define ROWS_PER_WAVE  16

__device__ __forceinline__ unsigned short f32_to_bf16_rne(float f) {
    unsigned int u = __float_as_uint(f);
    u += 0x7FFFu + ((u >> 16) & 1u);   // round-to-nearest-even
    return (unsigned short)(u >> 16);
}

union AFrag { v16bf v; unsigned short u[16]; };
union BFrag { v16bf v; uint4 q[2]; };

__device__ __forceinline__ void load_bfrag(const unsigned short* sW, int wr, int half,
                                           BFrag& b0, BFrag& b1) {
    // B layout (32x16, 16-bit): elems 0..7 -> K = half*8 + e ; elems 8..15 -> K = 16 + half*8 + (e-8)
    const uint4* p0 = reinterpret_cast<const uint4*>(&sW[wr * ROW_PAD + half * 8]);
    b0.q[0] = p0[0]; b0.q[1] = p0[2];
    const uint4* p1 = reinterpret_cast<const uint4*>(&sW[wr * ROW_PAD + 32 + half * 8]);
    b1.q[0] = p1[0]; b1.q[1] = p1[2];
}

__global__ __launch_bounds__(256)
void vq_argmin_wmma_kernel(const float* __restrict__ x,
                           const float* __restrict__ weight,
                           float* __restrict__ out) {
    __shared__ unsigned short sW[CHUNK * ROW_PAD];   // 36 KB bf16 codebook chunk
    __shared__ float          sNorm[CHUNK];          // 1 KB ||w||^2

    const int tid  = threadIdx.x;
    const int lane = tid & 31;
    const int wave = tid >> 5;
    const int nloc = lane & 15;   // column-in-tile (B/C) == row-in-tile (A)
    const int half = lane >> 4;   // wave half (K interleave for A, K split for B)

    const int row_base = (blockIdx.x * WAVES_PER_BLK + wave) * ROWS_PER_WAVE;

    // latents row n maps to x[b][c][hw]: b = n>>10, hw = n&1023 (NCHW -> NHWC view).
    // 16-row tiles never cross a batch boundary (1024 % 16 == 0).
    const int n_row = row_base + nloc;
    const int bimg  = n_row >> 10;
    const int hw    = n_row & 1023;
    const float* xb = x + (size_t)bimg * (EMB_DIM * 1024) + hw;

    // ---- Build both resident A fragments (K = 0..31 and 32..63), per ISA 16-bit
    // A layout: elems 0..7 -> K = half*8 + e ; elems 8..15 -> K = 16 + half*8 + (e-8)
    AFrag a[2];
    #pragma unroll
    for (int f = 0; f < 2; ++f) {
        #pragma unroll
        for (int e = 0; e < 16; ++e) {
            int c = f * 32 + ((e < 8) ? (half * 8 + e) : (16 + half * 8 + (e - 8)));
            a[f].u[e] = f32_to_bf16_rne(xb[(size_t)c * 1024]);
        }
    }

    float minv[8];
    int   mini[8];
    #pragma unroll
    for (int r = 0; r < 8; ++r) { minv[r] = 3.402823466e38f; mini[r] = 0; }

    for (int chunk = 0; chunk < NUM_CODES / CHUNK; ++chunk) {
        __syncthreads();
        // ---- Stage CHUNK codebook rows into LDS as bf16 + squared norms (1 row/thread)
        {
            const float* wrow = weight + (size_t)(chunk * CHUNK + tid) * EMB_DIM;
            float nrm = 0.f;
            unsigned short tmp[EMB_DIM];
            #pragma unroll
            for (int j = 0; j < EMB_DIM / 4; ++j) {
                float4 w4 = reinterpret_cast<const float4*>(wrow)[j];
                nrm += w4.x * w4.x + w4.y * w4.y + w4.z * w4.z + w4.w * w4.w;
                tmp[4 * j + 0] = f32_to_bf16_rne(w4.x);
                tmp[4 * j + 1] = f32_to_bf16_rne(w4.y);
                tmp[4 * j + 2] = f32_to_bf16_rne(w4.z);
                tmp[4 * j + 3] = f32_to_bf16_rne(w4.w);
            }
            uint4* dst = reinterpret_cast<uint4*>(&sW[tid * ROW_PAD]);
            #pragma unroll
            for (int j = 0; j < EMB_DIM / 8; ++j)
                dst[j] = reinterpret_cast<const uint4*>(tmp)[j];
            sNorm[tid] = nrm;
        }
        __syncthreads();

        // ---- 2 column-tiles (2x16 codes) per iteration: two independent WMMA
        // accumulator chains so the 2nd tile's WMMAs + LDS loads fill the 1st
        // tile's WMMA->VALU hazard window (4 coexec NOPs otherwise).
        for (int t = 0; t < CHUNK / 16; t += 2) {
            BFrag b00, b01, b10, b11;
            load_bfrag(sW, (t + 0) * 16 + nloc, half, b00, b01);
            load_bfrag(sW, (t + 1) * 16 + nloc, half, b10, b11);

            v8f acc0 = {}, acc1 = {};
            acc0 = __builtin_amdgcn_wmma_f32_16x16x32_bf16(false, a[0].v, false, b00.v,
                                                           (short)0, acc0, false, false);
            acc1 = __builtin_amdgcn_wmma_f32_16x16x32_bf16(false, a[0].v, false, b10.v,
                                                           (short)0, acc1, false, false);
            acc0 = __builtin_amdgcn_wmma_f32_16x16x32_bf16(false, a[1].v, false, b01.v,
                                                           (short)0, acc0, false, false);
            acc1 = __builtin_amdgcn_wmma_f32_16x16x32_bf16(false, a[1].v, false, b11.v,
                                                           (short)0, acc1, false, false);

            const int   code0 = chunk * CHUNK + (t + 0) * 16 + nloc;
            const int   code1 = chunk * CHUNK + (t + 1) * 16 + nloc;
            const float wn0   = sNorm[(t + 0) * 16 + nloc];
            const float wn1   = sNorm[(t + 1) * 16 + nloc];
            // C/D layout: lane holds column nloc, rows m = half*8 + r (r = VGPR index)
            #pragma unroll
            for (int r = 0; r < 8; ++r) {
                float s0 = wn0 - 2.0f * acc0[r];
                float s1 = wn1 - 2.0f * acc1[r];
                if (s0 < minv[r]) { minv[r] = s0; mini[r] = code0; }
                if (s1 < minv[r]) { minv[r] = s1; mini[r] = code1; }
            }
        }
    }

    // ---- Cross-lane argmin over the 16 columns held per lane-half (xor stays in half)
    #pragma unroll
    for (int r = 0; r < 8; ++r) {
        #pragma unroll
        for (int m = 1; m < 16; m <<= 1) {
            float ov = __shfl_xor(minv[r], m, 32);
            int   oi = __shfl_xor(mini[r], m, 32);
            if (ov < minv[r] || (ov == minv[r] && oi < mini[r])) {
                minv[r] = ov; mini[r] = oi;
            }
        }
    }

    // ---- Gather winning codebook rows (exact f32 values); 16 lanes x float4 per row
    #pragma unroll
    for (int r = 0; r < 8; ++r) {
        const int m   = half * 8 + r;
        const int idx = mini[r];   // identical across the 16 lanes of this half
        const float4 w4 =
            reinterpret_cast<const float4*>(weight + (size_t)idx * EMB_DIM)[nloc];
        reinterpret_cast<float4*>(out + (size_t)(row_base + m) * EMB_DIM)[nloc] = w4;
    }
}

extern "C" void kernel_launch(void* const* d_in, const int* in_sizes, int n_in,
                              void* d_out, int out_size, void* d_ws, size_t ws_size,
                              hipStream_t stream) {
    (void)in_sizes; (void)n_in; (void)out_size; (void)d_ws; (void)ws_size;
    const float* x = (const float*)d_in[0];      // [64, 64, 32, 32] f32
    const float* w = (const float*)d_in[1];      // [1024, 64] f32
    float* out = (float*)d_out;                  // [65536, 64] f32

    const int n_rows = 64 * 32 * 32;                              // 65536
    dim3 grid(n_rows / (WAVES_PER_BLK * ROWS_PER_WAVE));          // 512 blocks
    vq_argmin_wmma_kernel<<<grid, 256, 0, stream>>>(x, w, out);
}
